// GATLayer_75076028334882
// MI455X (gfx1250) — matrix operand order repaired and checked by reading, not compiled, not run on hardware
//
#include <hip/hip_runtime.h>
#include <hip/hip_bf16.h>
#include <stdint.h>

#define N_NODES 10000
#define N_EDGES 160000
#define EMB 128
#define HEADS 8
#define HC (HEADS * EMB)            // 1024
#define E_TOT (N_EDGES + N_NODES)   // 170000
#define M_TILES (N_NODES / 16)      // 625 (exact)
#define N_TILES (HC / 16)           // 64
#define NEG_SLOPE 0.2f
#define SOFTMAX_EPS 1e-16f

typedef __attribute__((ext_vector_type(16))) __bf16       v16bf;
typedef __attribute__((ext_vector_type(8)))  float        v8f;
typedef __attribute__((ext_vector_type(8)))  unsigned int v8u;

// ---------- helpers ----------
__device__ __forceinline__ unsigned short f32_to_bf16_rne(float f) {
    unsigned u = __float_as_uint(f);
    unsigned rounding = 0x7FFFu + ((u >> 16) & 1u);
    return (unsigned short)((u + rounding) >> 16);
}
// monotonic f32 -> u32 encoding so unsigned atomicMax == float max
__device__ __forceinline__ unsigned enc_f32(float f) {
    unsigned u = __float_as_uint(f);
    return (u & 0x80000000u) ? ~u : (u | 0x80000000u);
}
__device__ __forceinline__ float dec_f32(unsigned e) {
    unsigned u = (e & 0x80000000u) ? (e & 0x7FFFFFFFu) : ~e;
    return __uint_as_float(u);
}
__device__ __forceinline__ float lrelu(float v) { return v > 0.f ? v : NEG_SLOPE * v; }

// ---------- convert x to bf16 ----------
__global__ void __launch_bounds__(256) k_cvt_x(const float* __restrict__ x,
                                               unsigned short* __restrict__ xb, int n) {
    int i = blockIdx.x * blockDim.x + threadIdx.x;
    if (i < n) xb[i] = f32_to_bf16_rne(x[i]);
}

// ---------- pack W_l / W_r into WMMA B-matrix lane layout (bf16) ----------
// pW[((ntile*4+kstep)*32 + lane)*16 + i] = W[(kstep*32 + (lane>>4)*8 + (i&7) + (i>>3)*16)*HC + ntile*16 + (lane&15)]
__global__ void __launch_bounds__(256) k_pack_w(const float* __restrict__ Wl,
                                                const float* __restrict__ Wr,
                                                unsigned short* __restrict__ pWl,
                                                unsigned short* __restrict__ pWr) {
    int t = blockIdx.x * blockDim.x + threadIdx.x;   // 0 .. 2*131072-1
    if (t >= 2 * EMB * HC) return;
    int which = t >> 17;          // 131072 entries per matrix
    int u     = t & 131071;
    int i     = u & 15;
    int lane  = (u >> 4) & 31;
    int kstep = (u >> 9) & 3;
    int ntile = u >> 11;          // 0..63
    int k = kstep * 32 + (lane >> 4) * 8 + (i & 7) + (i >> 3) * 16;
    int n = ntile * 16 + (lane & 15);
    const float* W = which ? Wr : Wl;
    unsigned short* pW = which ? pWr : pWl;
    pW[u] = f32_to_bf16_rne(W[k * HC + n]);
}

// ---------- WMMA GEMM: out[10000,1024] = x(bf16) @ W(bf16-packed) + bias ----------
// one wave per 16x16 output tile, K=128 -> 4x v_wmma_f32_16x16x32_bf16
__global__ void __launch_bounds__(256) k_gemm(const unsigned short* __restrict__ xb,
                                              const unsigned short* __restrict__ pWl,
                                              const unsigned short* __restrict__ pWr,
                                              const float* __restrict__ b_l,
                                              const float* __restrict__ b_r,
                                              float* __restrict__ xl,
                                              float* __restrict__ xr) {
    const int lane  = threadIdx.x & 31;
    const int wave  = threadIdx.x >> 5;
    const int mtile = blockIdx.y * 8 + wave;
    if (mtile >= M_TILES) return;                       // wave-uniform exit
    const int ntile = blockIdx.x;
    const unsigned short* pW   = blockIdx.z ? pWr : pWl;
    const float*          bias = blockIdx.z ? b_r : b_l;
    float*                out  = blockIdx.z ? xr  : xl;

    const int m   = lane & 15;        // A row / D column index share lane&15 semantics
    const int kg  = lane >> 4;        // K-group half
    const int row = mtile * 16 + m;

    v8f acc = {0.f, 0.f, 0.f, 0.f, 0.f, 0.f, 0.f, 0.f};
#pragma unroll
    for (int ks = 0; ks < 4; ++ks) {
        // A: elements 0..7 -> K = ks*32 + kg*8 + [0..8); elements 8..15 -> +16
        const unsigned short* pa = xb + (size_t)row * EMB + ks * 32 + kg * 8;
        uint4 a0 = *reinterpret_cast<const uint4*>(pa);
        uint4 a1 = *reinterpret_cast<const uint4*>(pa + 16);
        v8u au = {a0.x, a0.y, a0.z, a0.w, a1.x, a1.y, a1.z, a1.w};
        v16bf A = __builtin_bit_cast(v16bf, au);
        // B: pre-packed contiguous 32B per lane
        const unsigned short* pb = pW + ((size_t)(ntile * 4 + ks) * 32 + lane) * 16;
        uint4 b0 = *reinterpret_cast<const uint4*>(pb);
        uint4 b1 = *reinterpret_cast<const uint4*>(pb + 8);
        v8u bu = {b0.x, b0.y, b0.z, b0.w, b1.x, b1.y, b1.z, b1.w};
        v16bf B = __builtin_bit_cast(v16bf, bu);
        acc = __builtin_amdgcn_wmma_f32_16x16x32_bf16(false, A, false, B,
                                                      (short)0, acc, false, false);
    }
    const int ncol = ntile * 16 + (lane & 15);
    const float bv = bias[ncol];
#pragma unroll
    for (int r = 0; r < 8; ++r) {
        int mrow = mtile * 16 + r + 8 * kg;             // C/D layout: M = r + 8*(lane/16)
        out[(size_t)mrow * HC + ncol] = acc[r] + bv;
    }
}

// ---------- per-edge attention score + segment max (one wave per edge) ----------
__global__ void __launch_bounds__(256) k_score(const int* __restrict__ ei,
                                               const float* __restrict__ xl,
                                               const float* __restrict__ xr,
                                               const float* __restrict__ att,
                                               float* __restrict__ score,
                                               unsigned* __restrict__ smax) {
    int gid  = blockIdx.x * blockDim.x + threadIdx.x;
    int e    = gid >> 5;
    int lane = gid & 31;
    if (e >= E_TOT) return;
    int src = (e < N_EDGES) ? ei[e] : (e - N_EDGES);
    int dst = (e < N_EDGES) ? ei[N_EDGES + e] : (e - N_EDGES);
    const float4* pl = reinterpret_cast<const float4*>(xl + (size_t)src * HC) + lane;
    const float4* pr = reinterpret_cast<const float4*>(xr + (size_t)dst * HC) + lane;
    const float4* pa = reinterpret_cast<const float4*>(att) + lane;
#pragma unroll
    for (int h = 0; h < HEADS; ++h) {
        float4 a  = pa[h * 32];
        float4 vl = pl[h * 32];
        float4 vr = pr[h * 32];
        float s = a.x * lrelu(vl.x + vr.x);
        s      += a.y * lrelu(vl.y + vr.y);
        s      += a.z * lrelu(vl.z + vr.z);
        s      += a.w * lrelu(vl.w + vr.w);
        s += __shfl_xor(s, 16, 32);
        s += __shfl_xor(s,  8, 32);
        s += __shfl_xor(s,  4, 32);
        s += __shfl_xor(s,  2, 32);
        s += __shfl_xor(s,  1, 32);
        if (lane == 0) {
            score[e * HEADS + h] = s;
            atomicMax(&smax[dst * HEADS + h], enc_f32(s));
        }
    }
}

// ---------- exp(score - max) + segment sum ----------
__global__ void __launch_bounds__(256) k_expv(const int* __restrict__ ei,
                                              float* __restrict__ sc,      // in: score, out: expv
                                              const unsigned* __restrict__ smax,
                                              float* __restrict__ ssum) {
    int t = blockIdx.x * blockDim.x + threadIdx.x;
    if (t >= E_TOT * HEADS) return;
    int e = t >> 3, h = t & 7;
    int dst = (e < N_EDGES) ? ei[N_EDGES + e] : (e - N_EDGES);
    float m  = dec_f32(smax[dst * HEADS + h]);
    float ev = __expf(sc[t] - m);
    sc[t] = ev;
    atomicAdd(&ssum[dst * HEADS + h], ev);
}

// ---------- alpha + weighted scatter-aggregation (one wave per edge) ----------
__global__ void __launch_bounds__(256) k_aggr(const int* __restrict__ ei,
                                              const float* __restrict__ xl,
                                              float* __restrict__ alpha,   // in: expv, out: alpha
                                              const float* __restrict__ ssum,
                                              float* __restrict__ aggr) {
    int gid  = blockIdx.x * blockDim.x + threadIdx.x;
    int e    = gid >> 5;
    int lane = gid & 31;
    if (e >= E_TOT) return;
    int src = (e < N_EDGES) ? ei[e] : (e - N_EDGES);
    int dst = (e < N_EDGES) ? ei[N_EDGES + e] : (e - N_EDGES);
    const float4* pl = reinterpret_cast<const float4*>(xl + (size_t)src * HC) + lane;
#pragma unroll
    for (int h = 0; h < HEADS; ++h) {
        float ev = alpha[e * HEADS + h];                         // wave-uniform
        float a  = ev / (ssum[dst * HEADS + h] + SOFTMAX_EPS);
        if (lane == 0) alpha[e * HEADS + h] = a;
        float4 vl = pl[h * 32];
        float* dp = aggr + (size_t)dst * HC + h * EMB + lane * 4;
        atomicAdd(dp + 0, vl.x * a);
        atomicAdd(dp + 1, vl.y * a);
        atomicAdd(dp + 2, vl.z * a);
        atomicAdd(dp + 3, vl.w * a);
    }
}

// ---------- head mean + bias ----------
__global__ void __launch_bounds__(256) k_final(const float* __restrict__ aggr,
                                               const float* __restrict__ bias,
                                               float* __restrict__ latent) {
    int t = blockIdx.x * blockDim.x + threadIdx.x;
    if (t >= N_NODES * EMB) return;
    int n = t >> 7, c = t & 127;
    float s = 0.f;
#pragma unroll
    for (int h = 0; h < HEADS; ++h) s += aggr[(size_t)n * HC + h * EMB + c];
    latent[t] = s * (1.f / HEADS) + bias[c];
}

// ---------- full_edge_index (as float, per harness output dtype) ----------
__global__ void __launch_bounds__(256) k_eidx(const int* __restrict__ ei,
                                              float* __restrict__ out) {
    int e = blockIdx.x * blockDim.x + threadIdx.x;
    if (e >= E_TOT) return;
    int s = (e < N_EDGES) ? ei[e] : (e - N_EDGES);
    int d = (e < N_EDGES) ? ei[N_EDGES + e] : (e - N_EDGES);
    out[e] = (float)s;
    out[E_TOT + e] = (float)d;
}

extern "C" void kernel_launch(void* const* d_in, const int* in_sizes, int n_in,
                              void* d_out, int out_size, void* d_ws, size_t ws_size,
                              hipStream_t stream) {
    const float* x    = (const float*)d_in[0];
    const float* W_l  = (const float*)d_in[1];
    const float* b_l  = (const float*)d_in[2];
    const float* W_r  = (const float*)d_in[3];
    const float* b_r  = (const float*)d_in[4];
    const float* att  = (const float*)d_in[5];
    const float* bias = (const float*)d_in[6];
    const int*   ei   = (const int*)d_in[7];

    // workspace carve-out (256B aligned)
    char* ws = (char*)d_ws;
    size_t off = 0;
    auto carve = [&](size_t bytes) {
        void* p = ws + off;
        off = (off + bytes + 255) & ~(size_t)255;
        return p;
    };
    unsigned short* xb   = (unsigned short*)carve((size_t)N_NODES * EMB * 2);
    unsigned short* pWl  = (unsigned short*)carve((size_t)EMB * HC * 2);
    unsigned short* pWr  = (unsigned short*)carve((size_t)EMB * HC * 2);
    float*          xl   = (float*)carve((size_t)N_NODES * HC * 4);
    float*          xr   = (float*)carve((size_t)N_NODES * HC * 4);
    unsigned*       smax = (unsigned*)carve((size_t)N_NODES * HEADS * 4);
    float*          ssum = (float*)carve((size_t)N_NODES * HEADS * 4);
    float*          aggr = (float*)carve((size_t)N_NODES * HC * 4);
    (void)ws_size; (void)in_sizes; (void)n_in; (void)out_size;

    // output regions (all f32, concatenated in return order)
    float* latent   = (float*)d_out;                      // N*EMB
    float* eidx_out = latent + (size_t)N_NODES * EMB;     // 2*E_TOT
    float* alpha    = eidx_out + (size_t)2 * E_TOT;       // E_TOT*HEADS (score->expv->alpha in place)

    hipMemsetAsync(smax, 0, (size_t)N_NODES * HEADS * 4, stream);   // encoded -inf
    hipMemsetAsync(ssum, 0, (size_t)N_NODES * HEADS * 4, stream);
    hipMemsetAsync(aggr, 0, (size_t)N_NODES * HC * 4, stream);

    k_cvt_x <<<(N_NODES * EMB + 255) / 256, 256, 0, stream>>>(x, xb, N_NODES * EMB);
    k_pack_w<<<(2 * EMB * HC + 255) / 256, 256, 0, stream>>>(W_l, W_r, pWl, pWr);
    k_gemm  <<<dim3(N_TILES, (M_TILES + 7) / 8, 2), 256, 0, stream>>>(xb, pWl, pWr, b_l, b_r, xl, xr);
    k_score <<<(E_TOT * 32 + 255) / 256, 256, 0, stream>>>(ei, xl, xr, att, alpha, smax);
    k_expv  <<<(E_TOT * HEADS + 255) / 256, 256, 0, stream>>>(ei, alpha, smax, ssum);
    k_aggr  <<<(E_TOT * 32 + 255) / 256, 256, 0, stream>>>(ei, xl, alpha, ssum, aggr);
    k_final <<<(N_NODES * EMB + 255) / 256, 256, 0, stream>>>(aggr, bias, latent);
    k_eidx  <<<(E_TOT + 255) / 256, 256, 0, stream>>>(ei, eidx_out);
}
